// FrozenPrefixCache_1760936591860
// MI455X (gfx1250) — matrix operand order repaired and checked by reading, not compiled, not run on hardware
//
#include <hip/hip_runtime.h>
#include <hip/hip_bf16.h>
#include <math.h>

// ---------------------------------------------------------------------------
// Problem constants (from reference): B=4, T=4096, DIM=768, MEM=192
// ---------------------------------------------------------------------------
static constexpr int BATCH = 4;
static constexpr int T     = 4096;
static constexpr int DIM   = 768;
static constexpr int MEM   = 192;

typedef __attribute__((ext_vector_type(16))) __bf16 v16bf;
typedef __attribute__((ext_vector_type(8)))  __bf16 v8bf;
typedef __attribute__((ext_vector_type(4)))  __bf16 v4bf;
typedef __attribute__((ext_vector_type(8)))  float  v8f;

// Tensor Data Mover availability (device pass; 6-arg builtin on this toolchain)
#if defined(__HIP_DEVICE_COMPILE__) && defined(__gfx1250__) && \
    __has_builtin(__builtin_amdgcn_tensor_load_to_lds) &&      \
    __has_builtin(__builtin_amdgcn_s_wait_tensorcnt)
#define USE_TDM 1
#else
#define USE_TDM 0
#endif

#if USE_TDM
typedef unsigned int v4u __attribute__((ext_vector_type(4)));
typedef int          v8i __attribute__((ext_vector_type(8)));
typedef int          v4i __attribute__((ext_vector_type(4)));

// LDS byte offset of a __shared__ object (AS3 pointer value == LDS offset)
static __device__ inline unsigned lds_off(const void* p) {
    return (unsigned)(unsigned long long)(__attribute__((address_space(3))) const void*)p;
}
#endif

// D = A(16x32 bf16) * B(32x16 bf16) + C(16x16 f32)
static __device__ inline v8f wmma_bf16(v16bf a, v16bf b, v8f c) {
    return __builtin_amdgcn_wmma_f32_16x16x32_bf16(
        /*neg_a=*/false, a, /*neg_b=*/false, b,
        /*c_mod=*/(short)0, c, /*reuse_a=*/false, /*reuse_b=*/false);
}

// A-fragment (contiguous-k order used here; B uses the same): lane holds row m = lane&15.
// Element order e=0..15 maps to k = h*16+e, h = lane>>4.
// => two contiguous 16B chunks at +h*16 and +h*16+8 (bf16 units).
static __device__ inline v16bf load_afrag(const __bf16* rowptr, int h) {
    v8bf lo = *(const v8bf*)(rowptr + h * 16);
    v8bf hi = *(const v8bf*)(rowptr + h * 16 + 8);
    v16bf a;
#pragma unroll
    for (int e = 0; e < 8; ++e) { a[e] = lo[e]; a[e + 8] = hi[e]; }
    return a;
}

static __device__ inline v8f zero8() {
    v8f z = {0.f, 0.f, 0.f, 0.f, 0.f, 0.f, 0.f, 0.f};
    return z;
}

// ---------------------------------------------------------------------------
// f32 -> bf16 conversion (4-wide)
// ---------------------------------------------------------------------------
__global__ void cvt4_kernel(const float* __restrict__ s, __bf16* __restrict__ d, int n4) {
    int i = blockIdx.x * blockDim.x + threadIdx.x;
    if (i < n4) {
        float4 f = ((const float4*)s)[i];
        v4bf o = {(__bf16)f.x, (__bf16)f.y, (__bf16)f.z, (__bf16)f.w};
        ((v4bf*)d)[i] = o;
    }
}

// transpose-convert: src[R][C] f32 -> dst[C][R] bf16
__global__ void cvtT_kernel(const float* __restrict__ s, __bf16* __restrict__ d, int R, int C) {
    int e = blockIdx.x * blockDim.x + threadIdx.x;
    if (e < R * C) {
        int r = e / C, c = e % C;
        d[(size_t)c * R + r] = (__bf16)s[e];
    }
}

// ---------------------------------------------------------------------------
// bf16 WMMA GEMM:  out[M][N] = A[M][K] * BT[N][K]^T + bias[N]
// One wave computes one 16x16 tile; K unrolled into a v_wmma chain.
// ---------------------------------------------------------------------------
template <int M, int N, int K, bool OUT_F32>
__global__ __launch_bounds__(256) void gemm_bf16_kernel(
    const __bf16* __restrict__ A, const __bf16* __restrict__ BT,
    const float* __restrict__ bias, void* __restrict__ out) {
    const int wave = threadIdx.x >> 5;
    const int lane = threadIdx.x & 31;
    const int tile = blockIdx.x * 8 + wave;
    constexpr int TN = N / 16;
    constexpr int TM = M / 16;
    if (tile >= TM * TN) return;           // no barriers in this kernel
    const int tm = tile / TN, tn = tile % TN;
    const int r = lane & 15, h = lane >> 4;

    const __bf16* arow = A + (size_t)(tm * 16 + r) * K;
    const __bf16* bcol = BT + (size_t)(tn * 16 + r) * K;

    v8f acc = zero8();
#pragma unroll
    for (int kk = 0; kk < K / 32; ++kk) {
        v16bf a = load_afrag(arow + kk * 32, h);
        v16bf b = *(const v16bf*)(bcol + kk * 32 + h * 16);  // B: contiguous K per lane
        acc = wmma_bf16(a, b, acc);
    }
    const float bv = bias[tn * 16 + r];
#pragma unroll
    for (int i = 0; i < 8; ++i) {
        const int row = tm * 16 + i + 8 * h;
        const int col = tn * 16 + r;
        const float v = acc[i] + bv;
        if (OUT_F32) ((float*)out)[(size_t)row * N + col] = v;
        else         ((__bf16*)out)[(size_t)row * N + col] = (__bf16)v;
    }
}

// ---------------------------------------------------------------------------
// Flash attention: 8 waves/WG, each wave owns 16 query rows (WG = 128 rows).
// K tile DMA'd into LDS by the Tensor Data Mover; V transposed at LDS fill;
// online softmax in registers.
// ---------------------------------------------------------------------------
static constexpr int KT = 32;           // keys per tile
static constexpr int KROW = MEM;        // 192 el = 384B rows: 32B aligned, TDM-contiguous
static constexpr int VPAD = KT + 16;    // 96B rows  -> 32B aligned vector reads
static constexpr int PPAD = KT + 8;     // 80B rows  -> 16B aligned vector reads

__global__ __launch_bounds__(256) void attn_kernel(
    const __bf16* __restrict__ qbf, const __bf16* __restrict__ kbf,
    const __bf16* __restrict__ vbf, const unsigned char* __restrict__ frozen,
    __bf16* __restrict__ ctxbf) {
    __shared__ __align__(32) __bf16 ldsK[KT][KROW];     // K tile, row-major [s][m]
    __shared__ __align__(32) __bf16 ldsVT[MEM][VPAD];   // V tile, transposed [m][s]
    __shared__ __align__(32) __bf16 ldsP[8][16][PPAD];  // per-wave P relayout C->A

    const int b = blockIdx.y;
    const int qblock = blockIdx.x * 128;
    const int tid = threadIdx.x;
    const int w = tid >> 5, lane = tid & 31;
    const int r = lane & 15, h = lane >> 4;
    const int qbase = qblock + w * 16;

    // Preload Q fragments (16 rows x MEM) -> 6 A-frags of K=32
    v16bf qa[6];
    const __bf16* qrow = qbf + (size_t)(b * T + qbase + r) * MEM;
#pragma unroll
    for (int kk = 0; kk < 6; ++kk) qa[kk] = load_afrag(qrow + kk * 32, h);

    v8f acc[12];
#pragma unroll
    for (int v = 0; v < 12; ++v) acc[v] = zero8();
    float m_run[8], l_run[8];
#pragma unroll
    for (int i = 0; i < 8; ++i) { m_run[i] = -INFINITY; l_run[i] = 0.f; }

    const int ktiles = (qblock + 128) / KT;   // causal: only tiles with s <= last row of WG
    const float scale = 0.07216878364870323f; // 1/sqrt(192)

    for (int kt = 0; kt < ktiles; ++kt) {
        const int sbase = kt * KT;
        __syncthreads();   // previous tile fully consumed before overwrite

#if USE_TDM
        // --- K tile via Tensor Data Mover: 2D descriptor, 32 rows x 192 bf16 ---
        if (w == 0) {
            const unsigned long long ga =
                (unsigned long long)(const void*)(kbf + (size_t)(b * T + sbase) * MEM);
            // D# group0: count=1 | lds_addr[63:32] | global_addr[120:64] | type=2
            v4u g0 = {1u, lds_off(&ldsK[0][0]),
                      (unsigned)ga, (unsigned)(ga >> 32) | (2u << 30)};
            // D# group1: data_size=2B; tensor_dim0=192, tensor_dim1=32;
            //            tile_dim0=192, tile_dim1=32; tensor_dim0_stride=192
            v8i g1 = {(int)0x00010000u,          // workgroup_mask=0, data_size=1(2B)
                      (int)(192u << 16),         // tensor_dim0[15:0] << 16
                      (int)(32u << 16),          // tensor_dim1[15:0] << 16
                      (int)(192u << 16),         // tile_dim0 << 16
                      32,                        // tile_dim1 (tile_dim2=0)
                      192,                       // tensor_dim0_stride[31:0]
                      0, 0};
            v4i gz4 = {0, 0, 0, 0};              // groups 2/3 unused (2D tensor)
            v8i gz8 = {0, 0, 0, 0, 0, 0, 0, 0};  // extra group (unused)
            __builtin_amdgcn_tensor_load_to_lds(g0, g1, gz4, gz4, gz8, 0);
        }
#else
        // --- K tile staged manually: 32 x 192 bf16 as 16B chunks ---
        for (int c = tid; c < KT * (MEM / 8); c += 256) {
            const int row = c / (MEM / 8);
            const int co  = (c % (MEM / 8)) * 8;
            *(v8bf*)&ldsK[row][co] =
                *(const v8bf*)(kbf + (size_t)(b * T + sbase + row) * MEM + co);
        }
#endif
        // Stage V tile transposed: [m][s] (overlaps with the TDM DMA)
        for (int e = tid; e < KT * MEM; e += 256) {
            const int s = e / MEM, m = e % MEM;
            ldsVT[m][s] = vbf[(size_t)(b * T + sbase + s) * MEM + m];
        }
        // gfx1250 prefetch of next K tile while this one is consumed
        if (kt + 1 < ktiles && tid < KT) {
            __builtin_prefetch(kbf + (size_t)(b * T + sbase + KT + tid) * MEM, 0, 1);
        }
#if USE_TDM
        if (w == 0) __builtin_amdgcn_s_wait_tensorcnt((short)0);
#endif
        __syncthreads();

        if (sbase <= qbase + 15) {   // tile intersects this wave's causal region
            // S(16x32) = Q * K^T : 2 n-frags x 6 k-steps
            v8f sfr[2];
#pragma unroll
            for (int nt = 0; nt < 2; ++nt) {
                v8f s_ = zero8();
#pragma unroll
                for (int kk = 0; kk < 6; ++kk) {
                    v16bf bf = *(const v16bf*)&ldsK[nt * 16 + r][kk * 32 + h * 16];
                    s_ = wmma_bf16(qa[kk], bf, s_);
                }
                sfr[nt] = s_;
            }
            // scale + causal/frozen mask + row max (C layout: row = i+8h, col = lane&15)
            float rmax[8];
#pragma unroll
            for (int i = 0; i < 8; ++i) {
                const int t_g = qbase + i + 8 * h;
                float best = -INFINITY;
#pragma unroll
                for (int nt = 0; nt < 2; ++nt) {
                    const int s_g = sbase + nt * 16 + r;
                    float v = sfr[nt][i] * scale;
                    const bool ok = (s_g <= t_g) && (frozen[(size_t)b * T + s_g] != 0);
                    v = ok ? v : -INFINITY;
                    sfr[nt][i] = v;
                    best = fmaxf(best, v);
                }
#pragma unroll
                for (int msk = 1; msk < 16; msk <<= 1)
                    best = fmaxf(best, __shfl_xor(best, msk, 32));
                rmax[i] = best;
            }
            // online softmax update; emit P into per-wave LDS for C->A relayout
#pragma unroll
            for (int i = 0; i < 8; ++i) {
                const float mnew  = fmaxf(m_run[i], rmax[i]);
                const bool  dead  = (mnew == -INFINITY);
                const float alpha = dead ? 1.f : __expf(m_run[i] - mnew);
                float rsum = 0.f;
#pragma unroll
                for (int nt = 0; nt < 2; ++nt) {
                    const float p = dead ? 0.f : __expf(sfr[nt][i] - mnew);
                    ldsP[w][i + 8 * h][nt * 16 + r] = (__bf16)p;
                    rsum += p;
                }
#pragma unroll
                for (int msk = 1; msk < 16; msk <<= 1)
                    rsum += __shfl_xor(rsum, msk, 32);
                l_run[i] = l_run[i] * alpha + rsum;
                m_run[i] = mnew;
#pragma unroll
                for (int v = 0; v < 12; ++v) acc[v][i] *= alpha;
            }
            // ctx(16x192) += P(16x32) * V(32x192): K=32 -> 1 wmma per n-tile
            v16bf pa = load_afrag(&ldsP[w][r][0], h);
#pragma unroll
            for (int v = 0; v < 12; ++v) {
                v16bf bf = *(const v16bf*)&ldsVT[v * 16 + r][h * 16];
                acc[v] = wmma_bf16(pa, bf, acc[v]);
            }
        }
    }
    // Normalize and store ctx as bf16 for the output projection GEMM
#pragma unroll
    for (int i = 0; i < 8; ++i) {
        const float inv = (l_run[i] > 0.f) ? 1.f / l_run[i] : 0.f;
        const int t_g = qbase + i + 8 * h;
#pragma unroll
        for (int v = 0; v < 12; ++v) {
            ctxbf[(size_t)(b * T + t_g) * MEM + v * 16 + r] = (__bf16)(acc[v][i] * inv);
        }
    }
}

// ---------------------------------------------------------------------------
// Host-side launch
// ---------------------------------------------------------------------------
extern "C" void kernel_launch(void* const* d_in, const int* in_sizes, int n_in,
                              void* d_out, int out_size, void* d_ws, size_t ws_size,
                              hipStream_t stream) {
    const float* mu      = (const float*)d_in[0];          // (B,T,DIM)
    const float* k_state = (const float*)d_in[1];          // (B,T,MEM)
    const float* v_state = (const float*)d_in[2];          // (B,T,MEM)
    const unsigned char* frozen = (const unsigned char*)d_in[3];  // (B,T) bool
    const float* wq = (const float*)d_in[4];               // (DIM,MEM)
    const float* bq = (const float*)d_in[5];               // (MEM,)
    const float* wo = (const float*)d_in[6];               // (MEM,DIM)
    const float* bo = (const float*)d_in[7];               // (DIM,)
    float* out = (float*)d_out;                            // (B,T,DIM)

    // Workspace layout (bf16 buffers, 256B-aligned offsets)
    const size_t N_MU = (size_t)BATCH * T * DIM;   // 12,582,912
    const size_t N_KV = (size_t)BATCH * T * MEM;   //  3,145,728
    char* ws = (char*)d_ws;
    __bf16* mu_bf  = (__bf16*)(ws);                                   // 25,165,824 B
    __bf16* k_bf   = (__bf16*)(ws + 25165824);                        //  6,291,456 B
    __bf16* v_bf   = (__bf16*)(ws + 31457280);                        //  6,291,456 B
    __bf16* q_bf   = (__bf16*)(ws + 37748736);                        //  6,291,456 B
    __bf16* ctx_bf = (__bf16*)(ws + 44040192);                        //  6,291,456 B
    __bf16* wqT_bf = (__bf16*)(ws + 50331648);                        //    294,912 B
    __bf16* woT_bf = (__bf16*)(ws + 50626560);                        //    294,912 B
    // total ~48.6 MB

    // 1) Precision conversion (f32 -> bf16); weights transposed for B-fragment loads
    {
        int n4 = (int)(N_MU / 4);
        cvt4_kernel<<<(n4 + 255) / 256, 256, 0, stream>>>(mu, mu_bf, n4);
        n4 = (int)(N_KV / 4);
        cvt4_kernel<<<(n4 + 255) / 256, 256, 0, stream>>>(k_state, k_bf, n4);
        cvt4_kernel<<<(n4 + 255) / 256, 256, 0, stream>>>(v_state, v_bf, n4);
        const int nw = DIM * MEM;
        cvtT_kernel<<<(nw + 255) / 256, 256, 0, stream>>>(wq, wqT_bf, DIM, MEM); // -> [MEM][DIM]
        cvtT_kernel<<<(nw + 255) / 256, 256, 0, stream>>>(wo, woT_bf, MEM, DIM); // -> [DIM][MEM]
    }

    // 2) q = mu @ wq + bq : (16384 x 192), K=768
    {
        constexpr int M = BATCH * T, N = MEM, K = DIM;
        const int tiles = (M / 16) * (N / 16);
        gemm_bf16_kernel<M, N, K, false>
            <<<(tiles + 7) / 8, 256, 0, stream>>>(mu_bf, wqT_bf, bq, (void*)q_bf);
    }

    // 3) Flash attention -> ctx (bf16)
    {
        dim3 grid(T / 128, BATCH);
        attn_kernel<<<grid, 256, 0, stream>>>(q_bf, k_bf, v_bf, frozen, ctx_bf);
    }

    // 4) out = ctx @ wo + bo : (16384 x 768), K=192
    {
        constexpr int M = BATCH * T, N = DIM, K = MEM;
        const int tiles = (M / 16) * (N / 16);
        gemm_bf16_kernel<M, N, K, true>
            <<<(tiles + 7) / 8, 256, 0, stream>>>(ctx_bf, woT_bf, bo, (void*)out);
    }
}
